// GMMModel_44856638439907
// MI455X (gfx1250) — matrix operand order repaired and checked by reading, not compiled, not run on hardware
//
#include <hip/hip_runtime.h>

// ---------------------------------------------------------------------------
// Grouped GEMM (MoE-style) for MI455X / gfx1250, wave32.
// fp32 -> (hi,lo) bf16 split, 3x V_WMMA_F32_16X16X32_BF16 per tile per K-step
// (hi*hi + hi*lo + lo*hi), f32 accumulate. ~fp32 accuracy at bf16 matrix rate.
//
// Fast path (ws_size >= 134MB): prep kernel pre-splits W into bf16 hi/lo in
// [e][n][k] layout in d_ws; main loop streams B tiles with
// global_load_async_to_lds_b128 (ASYNCcnt). Double-buffered LDS pipeline:
// async-copy + A-conversion for chunk i+1 overlap the WMMAs of chunk i.
// Fallback: fully fused conversion, same pipeline.
// ---------------------------------------------------------------------------

typedef __attribute__((ext_vector_type(16))) __bf16 v16bf;
typedef __attribute__((ext_vector_type(8)))  float  v8f;

constexpr int GK  = 2048;   // K
constexpr int GN  = 2048;   // N
constexpr int BM  = 128;    // block tile M
constexpr int BN  = 128;    // block tile N
constexpr int BK  = 32;     // K step (bf16 WMMA depth)
constexpr int LDT = 40;     // padded LDS row stride (ushorts)

// --- bf16 helpers (hardware RNE convert via __bf16 casts) -------------------
__device__ __forceinline__ unsigned short f2bf(float f) {
    __bf16 h = (__bf16)f;
    return __builtin_bit_cast(unsigned short, h);
}
__device__ __forceinline__ float bf2f(unsigned short h) {
    return __uint_as_float(((unsigned int)h) << 16);
}
// split two floats -> packed hi u32 and packed lo u32 (elem0 in low half)
__device__ __forceinline__ void split2(float a, float b, unsigned& hi, unsigned& lo) {
    unsigned short ha = f2bf(a), hb = f2bf(b);
    hi = (unsigned)ha | ((unsigned)hb << 16);
    unsigned short la = f2bf(a - bf2f(ha)), lb = f2bf(b - bf2f(hb));
    lo = (unsigned)la | ((unsigned)lb << 16);
}

union FragU { v16bf v; uint4 q[2]; };

// ---------------------------------------------------------------------------
// Prep: W [e][k][n] fp32  ->  ws hi/lo bf16 in [e][n][k] (transposed) layout.
// ---------------------------------------------------------------------------
__global__ __launch_bounds__(256)
void w_split_transpose_kernel(const float* __restrict__ w,
                              unsigned short* __restrict__ wsHi,
                              unsigned short* __restrict__ wsLo)
{
    __shared__ unsigned short tHi[64 * 72];   // [n][k], pad 8
    __shared__ unsigned short tLo[64 * 72];

    const int e  = blockIdx.z;
    const int k0 = blockIdx.y * 64;
    const int n0 = blockIdx.x * 64;
    const int t  = threadIdx.x;

    const float* wb = w + ((size_t)e * GK + k0) * GN + n0;

    #pragma unroll
    for (int p = 0; p < 4; ++p) {
        const int idx = p * 256 + t;
        const int r   = idx >> 4;           // k row 0..63
        const int c4  = (idx & 15) * 4;     // n col 0,4,..60
        const float4 v = *(const float4*)(wb + (size_t)r * GN + c4);
        const float f[4] = {v.x, v.y, v.z, v.w};
        #pragma unroll
        for (int j = 0; j < 4; ++j) {
            unsigned short h = f2bf(f[j]);
            tHi[(c4 + j) * 72 + r] = h;
            tLo[(c4 + j) * 72 + r] = f2bf(f[j] - bf2f(h));
        }
    }
    __syncthreads();

    unsigned short* oh = wsHi + ((size_t)e * GN + n0) * GK + k0;
    unsigned short* ol = wsLo + ((size_t)e * GN + n0) * GK + k0;
    #pragma unroll
    for (int p = 0; p < 2; ++p) {
        const int idx  = p * 256 + t;
        const int n    = idx >> 3;
        const int part = idx & 7;
        *(uint4*)(oh + (size_t)n * GK + part * 8) = *(const uint4*)&tHi[n * 72 + part * 8];
        *(uint4*)(ol + (size_t)n * GK + part * 8) = *(const uint4*)&tLo[n * 72 + part * 8];
    }
}

// ---------------------------------------------------------------------------
// Main grouped-GEMM kernel, double-buffered pipeline.
// ---------------------------------------------------------------------------
template <bool WSPATH>
__global__ __launch_bounds__(256)
void gmm_bf16x3_kernel(const float* __restrict__ x,
                       const float* __restrict__ w,
                       const unsigned short* __restrict__ wsHi,
                       const unsigned short* __restrict__ wsLo,
                       const long long* __restrict__ group_list,
                       float* __restrict__ out)
{
    __shared__ unsigned short sAhi[2][BM * LDT];
    __shared__ unsigned short sAlo[2][BM * LDT];
    __shared__ unsigned short sBhi[2][BN * LDT];
    __shared__ unsigned short sBlo[2][BN * LDT];

    const int e = blockIdx.z;
    const long long start = (e == 0) ? 0 : group_list[e - 1];
    const long long end   = group_list[e];
    const int rows  = (int)(end - start);
    const int mBase = blockIdx.y * BM;
    if (mBase >= rows) return;
    const int n0 = blockIdx.x * BN;

    const int t    = threadIdx.x;
    const int lane = t & 31;
    const int wave = t >> 5;
    const int waveM = (wave >> 1) * 32;
    const int waveN = (wave & 1) * 64;

    const float* wBase = w + (size_t)e * GK * GN;
    const unsigned short* wsHiE = wsHi + (size_t)e * GN * GK;   // [n][k]
    const unsigned short* wsLoE = wsLo + (size_t)e * GN * GK;

    // ---- staging helpers ---------------------------------------------------
    auto stageB = [&](int kcS, int buf) {
        if (WSPATH) {
            const unsigned dBhi = (unsigned)(uintptr_t)&sBhi[buf][0];
            const unsigned dBlo = (unsigned)(uintptr_t)&sBlo[buf][0];
            #pragma unroll
            for (int q = 0; q < 2; ++q) {
                const int idx  = q * 256 + t;      // 0..511
                const int n    = idx >> 2;         // 0..127
                const int part = idx & 3;          // 16B chunk of 64B row
                const unsigned short* gh = wsHiE + (size_t)(n0 + n) * GK + kcS + part * 8;
                const unsigned short* gl = wsLoE + (size_t)(n0 + n) * GK + kcS + part * 8;
                const unsigned dh = dBhi + (unsigned)((n * LDT + part * 8) * 2);
                const unsigned dl = dBlo + (unsigned)((n * LDT + part * 8) * 2);
                asm volatile("global_load_async_to_lds_b128 %0, %1, off"
                             :: "v"(dh), "v"(gh) : "memory");
                asm volatile("global_load_async_to_lds_b128 %0, %1, off"
                             :: "v"(dl), "v"(gl) : "memory");
            }
        } else {
            #pragma unroll
            for (int p = 0; p < 4; ++p) {
                const int idx = p * 256 + t;       // 0..1023 2x2 quads
                const int kq  = (idx >> 6) * 2;    // 0,2,..30
                const int nq  = (idx & 63) * 2;    // 0,2,..126
                const float2 v0 = *(const float2*)(wBase + (size_t)(kcS + kq)     * GN + n0 + nq);
                const float2 v1 = *(const float2*)(wBase + (size_t)(kcS + kq + 1) * GN + n0 + nq);
                unsigned h0, l0, h1, l1;
                split2(v0.x, v1.x, h0, l0);
                split2(v0.y, v1.y, h1, l1);
                *(unsigned*)&sBhi[buf][(nq)     * LDT + kq] = h0;
                *(unsigned*)&sBlo[buf][(nq)     * LDT + kq] = l0;
                *(unsigned*)&sBhi[buf][(nq + 1) * LDT + kq] = h1;
                *(unsigned*)&sBlo[buf][(nq + 1) * LDT + kq] = l1;
            }
        }
    };

    auto stageA = [&](int kcS, int buf) {
        #pragma unroll
        for (int p = 0; p < 4; ++p) {
            const int idx = p * 256 + t;
            const int r   = idx >> 3;              // 0..127
            const int c4  = (idx & 7) * 4;         // 0,4,..28
            float4 v = make_float4(0.f, 0.f, 0.f, 0.f);
            if (mBase + r < rows)
                v = *(const float4*)(x + (start + mBase + r) * (size_t)GK + kcS + c4);
            unsigned h0, l0, h1, l1;
            split2(v.x, v.y, h0, l0);
            split2(v.z, v.w, h1, l1);
            *(unsigned*)&sAhi[buf][r * LDT + c4]     = h0;
            *(unsigned*)&sAlo[buf][r * LDT + c4]     = l0;
            *(unsigned*)&sAhi[buf][r * LDT + c4 + 2] = h1;
            *(unsigned*)&sAlo[buf][r * LDT + c4 + 2] = l1;
        }
    };

    v8f acc[2][4] = {};

    // ---- prologue: stage chunk 0 into buffer 0 -----------------------------
    stageB(0, 0);
    stageA(0, 0);
    if (WSPATH) asm volatile("s_wait_asynccnt 0" ::: "memory");
    __syncthreads();

    const int fr = lane & 15;
    const int kh = (lane >> 4) * 8;
    const int kb = (lane >> 4) * 16;

    for (int kc = 0; kc < GK; kc += BK) {
        const int cur = (kc >> 5) & 1;
        const int nxt = cur ^ 1;
        const bool more = (kc + BK) < GK;

        // ---- stage chunk i+1 into the other buffer (overlaps WMMA below) --
        if (more) {
            stageB(kc + BK, nxt);
            stageA(kc + BK, nxt);
            if (kc + 2 * BK < GK) {   // distance-2 x prefetch
                const int r = t >> 3;
                if (mBase + r < rows)
                    __builtin_prefetch(x + (start + mBase + r) * (size_t)GK + kc + 2 * BK + (t & 7) * 4, 0, 1);
            }
        }

        // ---- fragments from current buffer --------------------------------
        // A (16x32): lanes 0-15 K{0..7,16..23}; lanes 16-31 K{8..15,24..31}
        v16bf aHi[2], aLo[2], bHi[4], bLo[4];
        #pragma unroll
        for (int mi = 0; mi < 2; ++mi) {
            const unsigned short* ph = &sAhi[cur][(waveM + mi * 16 + fr) * LDT];
            const unsigned short* pl = &sAlo[cur][(waveM + mi * 16 + fr) * LDT];
            FragU uh, ul;
            uh.q[0] = *(const uint4*)(ph + kh);
            uh.q[1] = *(const uint4*)(ph + 16 + kh);
            ul.q[0] = *(const uint4*)(pl + kh);
            ul.q[1] = *(const uint4*)(pl + 16 + kh);
            aHi[mi] = uh.v;  aLo[mi] = ul.v;
        }
        // B (32x16): lanes 0-15 K{0..7,16..23}; lanes 16-31 K{8..15,24..31}
        #pragma unroll
        for (int ni = 0; ni < 4; ++ni) {
            const unsigned short* ph = &sBhi[cur][(waveN + ni * 16 + fr) * LDT] + kh;
            const unsigned short* pl = &sBlo[cur][(waveN + ni * 16 + fr) * LDT] + kh;
            FragU uh, ul;
            uh.q[0] = *(const uint4*)(ph);
            uh.q[1] = *(const uint4*)(ph + 16);
            ul.q[0] = *(const uint4*)(pl);
            ul.q[1] = *(const uint4*)(pl + 16);
            bHi[ni] = uh.v;  bLo[ni] = ul.v;
        }

        // ---- bf16x3 WMMA: hi*hi + hi*lo + lo*hi ---------------------------
        #pragma unroll
        for (int mi = 0; mi < 2; ++mi) {
            #pragma unroll
            for (int ni = 0; ni < 4; ++ni) {
                acc[mi][ni] = __builtin_amdgcn_wmma_f32_16x16x32_bf16(
                    false, aHi[mi], false, bHi[ni], (short)0, acc[mi][ni], false, false);
                acc[mi][ni] = __builtin_amdgcn_wmma_f32_16x16x32_bf16(
                    false, aHi[mi], false, bLo[ni], (short)0, acc[mi][ni], false, false);
                acc[mi][ni] = __builtin_amdgcn_wmma_f32_16x16x32_bf16(
                    false, aLo[mi], false, bHi[ni], (short)0, acc[mi][ni], false, false);
            }
        }

        // ---- handoff: next buffer fully resident, flip ---------------------
        if (more) {
            if (WSPATH) asm volatile("s_wait_asynccnt 0" ::: "memory");
            __syncthreads();
        }
    }

    // ---- epilogue: C layout — VGPR r: lanes 0-15 M=r, lanes 16-31 M=r+8 ---
    const int colL  = lane & 15;
    const int rhalf = (lane >> 4) * 8;
    #pragma unroll
    for (int mi = 0; mi < 2; ++mi) {
        const int mb = mBase + waveM + mi * 16 + rhalf;
        #pragma unroll
        for (int ni = 0; ni < 4; ++ni) {
            const long long c = n0 + waveN + ni * 16 + colL;
            #pragma unroll
            for (int r = 0; r < 8; ++r) {
                const int m = mb + r;
                if (m < rows)
                    out[(start + m) * (size_t)GN + c] = acc[mi][ni][r];
            }
        }
    }
}

extern "C" void kernel_launch(void* const* d_in, const int* in_sizes, int n_in,
                              void* d_out, int out_size, void* d_ws, size_t ws_size,
                              hipStream_t stream) {
    const float*     x  = (const float*)d_in[0];
    const float*     w  = (const float*)d_in[1];
    const long long* gl = (const long long*)d_in[2];
    // d_in[3] = group_type (0), d_in[4] = group_list_type (0) — hardcoded.

    const int E = in_sizes[2];
    const int T = in_sizes[0] / GK;

    dim3 block(256);
    dim3 grid(GN / BN, (T + BM - 1) / BM, E);

    const size_t perArr = (size_t)E * GK * GN * sizeof(unsigned short);
    if (ws_size >= 2 * perArr) {
        unsigned short* wsHi = (unsigned short*)d_ws;
        unsigned short* wsLo = wsHi + (size_t)E * GK * GN;
        dim3 pgrid(GN / 64, GK / 64, E);
        w_split_transpose_kernel<<<pgrid, block, 0, stream>>>(w, wsHi, wsLo);
        gmm_bf16x3_kernel<true><<<grid, block, 0, stream>>>(x, w, wsHi, wsLo, gl, (float*)d_out);
    } else {
        gmm_bf16x3_kernel<false><<<grid, block, 0, stream>>>(x, w, nullptr, nullptr, gl, (float*)d_out);
    }
}
